// Drifting_50337016709322
// MI455X (gfx1250) — matrix operand order repaired and checked by reading, not compiled, not run on hardware
//
#include <hip/hip_runtime.h>

// ---------------------------------------------------------------------------
// Problem: G=4096 gen rows, P=4096 pos rows, D=512, T=G+P=8192 targets.
//   K[i,j]   = exp(-||gen_i - tgt_j|| / 0.05), K[i,i]=0
//   nk[i,j]  = K[i,j] / sqrt(rowsum_i * colsum_j)
//   V[i,:]   = s_neg_i * (nk_pos @ pos)[i,:] - s_pos_i * (nk_neg @ gen)[i,:]
// Strategy: bf16 WMMA everywhere; tgt (row-major) + tgtT (feature-major) in
// ws (16 MB, L2-resident on MI455X's 192 MB L2); K recomputed, never stored.
// Pass 3 split into neg/pos halves across blockIdx.y for 2x grid parallelism
// and one accumulator set per wave; tiny combine kernel at the end.
// ---------------------------------------------------------------------------

typedef __attribute__((ext_vector_type(16))) __bf16 v16bf;
typedef __attribute__((ext_vector_type(8)))  float  v8f;
typedef __attribute__((ext_vector_type(4)))  unsigned int u32x4;

#define GN   4096
#define TN   8192
#define DN   512
#define INV_TEMP 20.0f

__device__ __forceinline__ v16bf load_frag16(const __bf16* p0, const __bf16* p1) {
    union { u32x4 u[2]; v16bf v; } cvt;
    cvt.u[0] = *(const u32x4*)p0;   // 8 bf16
    cvt.u[1] = *(const u32x4*)p1;   // 8 bf16
    return cvt.v;
}

__device__ __forceinline__ v8f wmma_bf16(v16bf a, v16bf b, v8f c) {
    // D(16x16 f32) = A(16x32 bf16) * B(32x16 bf16) + C
    return __builtin_amdgcn_wmma_f32_16x16x32_bf16(
        /*neg_a=*/false, a, /*neg_b=*/false, b,
        /*c_mod=*/(short)0, c, /*reuse_a=*/false, /*reuse_b=*/false);
}

// ---------------------------------------------------------------------------
// Kernel 0: zero row_sum (4096) + col_sum (8192) (contiguous 12288 floats)
// ---------------------------------------------------------------------------
__global__ void zero_kernel(float* p, int n) {
    int i = blockIdx.x * blockDim.x + threadIdx.x;
    if (i < n) p[i] = 0.0f;
}

// ---------------------------------------------------------------------------
// Kernel 1: per-row sum of squares + f32 -> bf16 conversion (row-major and
// transposed copies). One block of 128 threads per target row.
// ---------------------------------------------------------------------------
__global__ __launch_bounds__(128) void prep_kernel(
    const float* __restrict__ gen, const float* __restrict__ pos,
    __bf16* __restrict__ tgt, __bf16* __restrict__ tgtT,
    float* __restrict__ sq) {
    const int row = blockIdx.x;
    const float* src = (row < GN) ? (gen + (size_t)row * DN)
                                  : (pos + (size_t)(row - GN) * DN);
    float ss = 0.0f;
#pragma unroll
    for (int k = 0; k < 4; ++k) {
        const int f = threadIdx.x + k * 128;
        const float v = src[f];
        ss += v * v;
        const __bf16 b = (__bf16)v;
        tgt [(size_t)row * DN + f] = b;
        tgtT[(size_t)f * TN + row] = b;
    }
#pragma unroll
    for (int m = 16; m >= 1; m >>= 1) ss += __shfl_xor(ss, m, 32);
    __shared__ float partial[4];
    if ((threadIdx.x & 31) == 0) partial[threadIdx.x >> 5] = ss;
    __syncthreads();
    if (threadIdx.x == 0)
        sq[row] = partial[0] + partial[1] + partial[2] + partial[3];
}

// ---------------------------------------------------------------------------
// Kernel 2: stats pass. Grid (32 i-blocks, 64 j-blocks), 256 threads = 8
// waves; each wave owns a 16x128 strip (8 WMMA tiles), K-loop over D=512.
// Epilogue: K=exp(-dist/T), diag->0, atomic row/col sums.
// C-layout: VGPR r, lanes 0-15: (M=r, N=lane); lanes 16-31: (M=8+r, N=lane-16)
// ---------------------------------------------------------------------------
__global__ __launch_bounds__(256) void stats_kernel(
    const __bf16* __restrict__ tgt, const __bf16* __restrict__ tgtT,
    const float* __restrict__ sq,
    float* __restrict__ row_sum, float* __restrict__ col_sum) {
    const int wave = threadIdx.x >> 5;
    const int lane = threadIdx.x & 31;
    const int i0 = blockIdx.x * 128 + wave * 16;
    const int j0 = blockIdx.y * 128;

    v8f acc[8];
#pragma unroll
    for (int t = 0; t < 8; ++t) acc[t] = (v8f){0,0,0,0,0,0,0,0};

    // A fragment addressing (16x32 bf16): lane&15 = M row; lanes<16 hold
    // K {0..7,16..23}, lanes>=16 hold K {8..15,24..31}.
    const int arow = i0 + (lane & 15);
    const int koff = (lane < 16) ? 0 : 8;
    const __bf16* abase = tgt + (size_t)arow * DN;

    for (int ks = 0; ks < 16; ++ks) {
        const int d0 = ks * 32;
        const v16bf afrag = load_frag16(abase + d0 + koff, abase + d0 + koff + 16);
        // B fragment (32x16 bf16): lane&15 = N column; K order same as A, from tgt
        const __bf16* bt = tgt + (size_t)(j0 + (lane & 15)) * DN + d0 + koff;
#pragma unroll
        for (int t = 0; t < 8; ++t) {
            const v16bf bfrag = load_frag16(bt + (size_t)t * 16 * DN, bt + (size_t)t * 16 * DN + 16);
            acc[t] = wmma_bf16(afrag, bfrag, acc[t]);
        }
    }

    const int mbase = (lane < 16) ? 0 : 8;
    const int nl = lane & 15;
    float sqi[8], rowacc[8];
#pragma unroll
    for (int r = 0; r < 8; ++r) { sqi[r] = sq[i0 + mbase + r]; rowacc[r] = 0.0f; }

#pragma unroll
    for (int t = 0; t < 8; ++t) {
        const int j = j0 + t * 16 + nl;
        const float sqj = sq[j];
        float csum = 0.0f;
#pragma unroll
        for (int r = 0; r < 8; ++r) {
            const float d2 = sqi[r] + sqj - 2.0f * acc[t][r];
            const float dist = sqrtf(fmaxf(d2, 0.0f));
            float kv = __expf(-INV_TEMP * dist);
            if (j == i0 + mbase + r) kv = 0.0f;   // diagonal -> exp(-2e7) = 0
            csum += kv;
            rowacc[r] += kv;
        }
        csum += __shfl_xor(csum, 16, 32);          // fold M 0-7 with 8-15
        if (lane < 16) atomicAdd(&col_sum[j], csum);
    }
#pragma unroll
    for (int r = 0; r < 8; ++r) {
        float v = rowacc[r];
        v += __shfl_xor(v, 1, 32); v += __shfl_xor(v, 2, 32);
        v += __shfl_xor(v, 4, 32); v += __shfl_xor(v, 8, 32);
        if (nl == 0) atomicAdd(&row_sum[i0 + mbase + r], v);
    }
}

// ---------------------------------------------------------------------------
// Kernel 3: fused second pass (flash-attention style), one j-half per
// blockIdx.y (0 = neg/gen, 1 = pos/pos). Grid (256, 2), 256 threads.
// Per 256-j slab: wave w computes nk tiles for j-subtiles {w, w+8}
// (matmul1, 2 independent WMMA chains) -> bf16 LDS; after barrier all 8
// waves consume nk from LDS as A of matmul2 over their 64-feature slice.
// Block-owned outputs: vbuf[i0..i0+15][0..511] and s_arr[i0..i0+15].
// ---------------------------------------------------------------------------
__global__ __launch_bounds__(256) void fused_kernel(
    const __bf16* __restrict__ tgt, const __bf16* __restrict__ tgtT,
    const float* __restrict__ sq,
    const float* __restrict__ row_sum, const float* __restrict__ col_sum,
    float* __restrict__ vneg_buf, float* __restrict__ vpos_buf,
    float* __restrict__ s_neg, float* __restrict__ s_pos) {
    __shared__ __align__(16) __bf16 nk_lds[16][264];   // 256 + 8 pad, rows 16B-aligned
    __shared__ float s_lds[16];

    const int wave = threadIdx.x >> 5;
    const int lane = threadIdx.x & 31;
    const int i0 = blockIdx.x * 16;
    const int half = blockIdx.y;                 // 0: j in [0,4096), 1: [4096,8192)
    float* __restrict__ vbuf = half ? vpos_buf : vneg_buf;
    float* __restrict__ sarr = half ? s_pos : s_neg;

    if (threadIdx.x < 16) s_lds[threadIdx.x] = 0.0f;
    __syncthreads();

    const int mbase = (lane < 16) ? 0 : 8;
    const int nl = lane & 15;
    const int koff = (lane < 16) ? 0 : 8;

    float sqi[8], rsi[8], sacc[8];
#pragma unroll
    for (int r = 0; r < 8; ++r) {
        sqi[r] = sq[i0 + mbase + r];
        rsi[r] = row_sum[i0 + mbase + r];
        sacc[r] = 0.0f;
    }
    v8f vacc[4];
#pragma unroll
    for (int t = 0; t < 4; ++t) vacc[t] = (v8f){0,0,0,0,0,0,0,0};

    const __bf16* abase = tgt + (size_t)(i0 + nl) * DN;  // gen rows (same all waves)

    for (int sb = 0; sb < 16; ++sb) {                    // 16 slabs of 256 j
        const int jbase = half * GN + sb * 256;

        // ---- matmul1: two 16x16 nk tiles per wave (jtiles w and w+8) ----
        v8f acc0 = (v8f){0,0,0,0,0,0,0,0};
        v8f acc1 = (v8f){0,0,0,0,0,0,0,0};
        for (int ks = 0; ks < 16; ++ks) {
            const int d0 = ks * 32;
            const v16bf afrag = load_frag16(abase + d0 + koff, abase + d0 + koff + 16);
            const __bf16* bt = tgt + (size_t)(jbase + wave * 16 + nl) * DN + d0 + koff;
            const v16bf b0 = load_frag16(bt, bt + 16);
            const v16bf b1 = load_frag16(bt + (size_t)128 * DN, bt + (size_t)128 * DN + 16);
            acc0 = wmma_bf16(afrag, b0, acc0);
            acc1 = wmma_bf16(afrag, b1, acc1);
        }

        // ---- epilogue: exp + normalize, write bf16 nk to LDS, row sums ----
#pragma unroll
        for (int t = 0; t < 2; ++t) {
            const v8f acc = t ? acc1 : acc0;
            const int jt = wave * 16 + t * 128;          // offset within slab
            const int j = jbase + jt + nl;               // global target index
            const float sqj = sq[j];
            const float csj = col_sum[j];
#pragma unroll
            for (int r = 0; r < 8; ++r) {
                const int gi = i0 + mbase + r;
                const float d2 = sqi[r] + sqj - 2.0f * acc[r];
                const float dist = sqrtf(fmaxf(d2, 0.0f));
                float kv = __expf(-INV_TEMP * dist);
                if (j == gi) kv = 0.0f;                  // diagonal (neg half only)
                const float nk = kv * rsqrtf(fmaxf(rsi[r] * csj, 1e-12f));
                nk_lds[mbase + r][jt + nl] = (__bf16)nk;
                float v = nk;
                v += __shfl_xor(v, 1, 32); v += __shfl_xor(v, 2, 32);
                v += __shfl_xor(v, 4, 32); v += __shfl_xor(v, 8, 32);
                sacc[r] += v;                            // valid on lanes nl==0
            }
        }
        __syncthreads();

        // ---- matmul2: V slice += nk(16x256) @ tgt[jbase:+256, f0:f0+64] ----
#pragma unroll
        for (int kt = 0; kt < 8; ++kt) {
            const __bf16* ap = &nk_lds[nl][kt * 32 + koff];
            const v16bf afrag = load_frag16(ap, ap + 16);
            const __bf16* brow = tgtT + (size_t)(wave * 64 + nl) * TN + jbase + kt * 32 + koff;
#pragma unroll
            for (int nt = 0; nt < 4; ++nt) {
                const v16bf bfrag = load_frag16(brow + (size_t)nt * 16 * TN, brow + (size_t)nt * 16 * TN + 16);
                vacc[nt] = wmma_bf16(afrag, bfrag, vacc[nt]);
            }
        }
        __syncthreads();   // nk_lds consumed before next slab overwrites it
    }

    // ---- combine per-wave row-sum partials across the 8 waves ----
    if (nl == 0) {
#pragma unroll
        for (int r = 0; r < 8; ++r) atomicAdd(&s_lds[mbase + r], sacc[r]);
    }
    __syncthreads();
    if (threadIdx.x < 16) sarr[i0 + threadIdx.x] = s_lds[threadIdx.x];

    // ---- store block-owned V partial ----
#pragma unroll
    for (int r = 0; r < 8; ++r) {
        const int m = mbase + r;
#pragma unroll
        for (int nt = 0; nt < 4; ++nt) {
            const int f = wave * 64 + nt * 16 + nl;
            vbuf[(size_t)(i0 + m) * DN + f] = vacc[nt][r];
        }
    }
}

// ---------------------------------------------------------------------------
// Kernel 4: V = s_neg[i] * Vpos[i,f] - s_pos[i] * Vneg[i,f]
// ---------------------------------------------------------------------------
__global__ __launch_bounds__(256) void combine_kernel(
    const float* __restrict__ vneg_buf, const float* __restrict__ vpos_buf,
    const float* __restrict__ s_neg, const float* __restrict__ s_pos,
    float* __restrict__ out) {
    const int idx = blockIdx.x * blockDim.x + threadIdx.x;
    const int i = idx >> 9;            // / DN
    out[idx] = s_neg[i] * vpos_buf[idx] - s_pos[i] * vneg_buf[idx];
}

// ---------------------------------------------------------------------------
extern "C" void kernel_launch(void* const* d_in, const int* in_sizes, int n_in,
                              void* d_out, int out_size, void* d_ws, size_t ws_size,
                              hipStream_t stream) {
    const float* gen = (const float*)d_in[0];
    const float* pos = (const float*)d_in[1];
    float* out = (float*)d_out;

    // ws layout (~32.2 MB):
    //   tgt bf16 [8192][512] | tgtT bf16 [512][8192] | sq[8192] |
    //   row_sum[4096] | col_sum[8192] | s_neg[4096] | s_pos[4096] |
    //   vneg[4096*512] f32 | vpos[4096*512] f32
    __bf16* tgt  = (__bf16*)d_ws;
    __bf16* tgtT = tgt + (size_t)TN * DN;
    float*  sq   = (float*)(tgtT + (size_t)DN * TN);
    float*  row_sum = sq + TN;
    float*  col_sum = row_sum + GN;
    float*  s_neg = col_sum + TN;
    float*  s_pos = s_neg + GN;
    float*  vneg_buf = s_pos + GN;
    float*  vpos_buf = vneg_buf + (size_t)GN * DN;

    prep_kernel<<<TN, 128, 0, stream>>>(gen, pos, tgt, tgtT, sq);
    zero_kernel<<<(GN + TN + 255) / 256, 256, 0, stream>>>(row_sum, GN + TN);
    stats_kernel<<<dim3(GN / 128, TN / 128), 256, 0, stream>>>(
        tgt, tgtT, sq, row_sum, col_sum);
    fused_kernel<<<dim3(GN / 16, 2), 256, 0, stream>>>(
        tgt, tgtT, sq, row_sum, col_sum, vneg_buf, vpos_buf, s_neg, s_pos);
    combine_kernel<<<(GN * DN) / 256, 256, 0, stream>>>(
        vneg_buf, vpos_buf, s_neg, s_pos, out);
}
